// Attention_39926015984130
// MI455X (gfx1250) — compile-verified, matrix operand order checked by reading, not run on hardware
//
#include <hip/hip_runtime.h>

typedef __bf16 bf16;
typedef __attribute__((ext_vector_type(16))) __bf16 v16bf;
typedef __attribute__((ext_vector_type(8)))  __bf16 v8bf;
typedef __attribute__((ext_vector_type(8)))  float  v8f;

#define WMMA_BF16(a, b, c) \
  __builtin_amdgcn_wmma_f32_16x16x32_bf16(false, (a), false, (b), (short)0, (c), false, false)

constexpr int D_MODEL  = 1024;
constexpr int N_HEADS  = 16;
constexpr int HEAD_DIM = 64;
constexpr int SEQ      = 2048;
constexpr int BATCH    = 4;
constexpr int NTOK     = BATCH * SEQ;                       // 8192
constexpr size_t HSZ   = (size_t)NTOK * D_MODEL;            // elems per Q/K/V buffer

// ---------------------------------------------------------------------------
// CDNA5 async global->LDS (ASYNCcnt path). LDS address = low 32 bits of the
// generic pointer (aperture-based addrspacecast puts the LDS offset there).
// ---------------------------------------------------------------------------
__device__ inline void async_load_b128(void* lds_dst, const void* gsrc) {
  unsigned lds = (unsigned)(size_t)lds_dst;
  unsigned long long ga = (unsigned long long)(size_t)gsrc;
  asm volatile("global_load_async_to_lds_b128 %0, %1, off"
               :: "v"(lds), "v"(ga) : "memory");
}
__device__ inline void wait_async0() {
  asm volatile("s_wait_asynccnt 0" ::: "memory");
}
__device__ inline void wait_ds0() {
  asm volatile("s_wait_dscnt 0" ::: "memory");
}

// ---------------------------------------------------------------------------
// fp32 -> bf16 cast (bandwidth trivial)
// ---------------------------------------------------------------------------
__global__ void cast_f32_bf16(const float* __restrict__ in, bf16* __restrict__ out, int n) {
  int i = blockIdx.x * blockDim.x + threadIdx.x;
  int stride = gridDim.x * blockDim.x;
  for (; i < n; i += stride) out[i] = (bf16)in[i];
}

// Build a 16x32 bf16 A/B fragment for one lane: two contiguous 16-byte chunks.
__device__ inline v16bf load_frag2(const bf16* p) {
  v8bf lo = *(const v8bf*)p;
  v8bf hi = *(const v8bf*)(p + 16);
  v16bf r;
#pragma unroll
  for (int j = 0; j < 8; ++j) { r[j] = lo[j]; r[8 + j] = hi[j]; }
  return r;
}

// ---------------------------------------------------------------------------
// Block GEMM: C[M,N] = A[M,K] (row-major bf16) x B[K,N] (row-major bf16)
// 256 threads = 8 waves. Block tile 128x128, K-step 32. Wave tile 32x64.
// Double-buffered LDS: A tile via async DMA, B tile transposed via registers;
// next tile's DMA + global loads are in flight under the current WMMAs.
// ---------------------------------------------------------------------------
template <typename Epi>
__global__ __launch_bounds__(256)
void gemm_bf16_128x128(const bf16* __restrict__ A, const bf16* __restrict__ B,
                       int M, int N, int K, Epi epi) {
  constexpr int BM = 128, BN = 128, BK = 32, LDT = BK + 8;   // padded LDS stride
  __shared__ bf16 As[2][BM * LDT];   // [m][k]
  __shared__ bf16 Bt[2][BN * LDT];   // [n][k]  (B stored transposed)

  const int tid  = threadIdx.x;
  const int lane = tid & 31;
  const int wid  = tid >> 5;

  const int nTilesN = N / BN;
  const int bm = (blockIdx.x / nTilesN) * BM;
  const int bn = (blockIdx.x % nTilesN) * BN;

  const int waveM = (wid & 3) * 32;     // 4 waves over M
  const int waveN = (wid >> 2) * 64;    // 2 waves over N

  v8f acc[2][4];
#pragma unroll
  for (int mf = 0; mf < 2; ++mf)
#pragma unroll
    for (int nf = 0; nf < 4; ++nf)
#pragma unroll
      for (int j = 0; j < 8; ++j) acc[mf][nf][j] = 0.0f;

  const int ar  = lane & 15;
  const int klo = (lane & 16) ? 8 : 0;  // 16-bit A-frag K-chunk select per lane half

  const int arow = tid >> 1, ahalf = (tid & 1) * 16;   // A staging map
  const int br = tid >> 3, bcs = (tid & 7) * 16;       // B staging map

  const bf16* aSrcBase = A + (size_t)(bm + arow) * K + ahalf;
  const bf16* bSrcBase = B + (size_t)br * N + bn + bcs;

  // prologue: stage tile 0 into buffer 0
  async_load_b128(&As[0][arow * LDT + ahalf],     aSrcBase);
  async_load_b128(&As[0][arow * LDT + ahalf + 8], aSrcBase + 8);
  {
    v8bf lo = *(const v8bf*)bSrcBase;
    v8bf hi = *(const v8bf*)(bSrcBase + 8);
#pragma unroll
    for (int j = 0; j < 8; ++j) Bt[0][(bcs + j) * LDT + br]     = lo[j];
#pragma unroll
    for (int j = 0; j < 8; ++j) Bt[0][(bcs + 8 + j) * LDT + br] = hi[j];
  }
  wait_async0();
  __syncthreads();

  const int nk = K / BK;
  for (int i = 0; i < nk; ++i) {
    const int c = i & 1;
    const bool more = (i + 1) < nk;

    // issue next A tile DMA + next B tile global loads (latency under WMMA)
    v8bf blo, bhi;
    if (more) {
      const bf16* asrc = aSrcBase + (size_t)(i + 1) * BK;
      async_load_b128(&As[1 - c][arow * LDT + ahalf],     asrc);
      async_load_b128(&As[1 - c][arow * LDT + ahalf + 8], asrc + 8);
      const bf16* bsrc = bSrcBase + (size_t)(i + 1) * BK * N;
      blo = *(const v8bf*)bsrc;
      bhi = *(const v8bf*)(bsrc + 8);
      __builtin_prefetch(asrc + BK, 0, 1);   // global_prefetch_b8 two tiles ahead
    }

    // compute on current buffer
    v16bf aF[2], bF[4];
#pragma unroll
    for (int mf = 0; mf < 2; ++mf)
      aF[mf] = load_frag2(&As[c][(waveM + mf * 16 + ar) * LDT + klo]);
#pragma unroll
    for (int nf = 0; nf < 4; ++nf)
      bF[nf] = load_frag2(&Bt[c][(waveN + nf * 16 + ar) * LDT + klo]);

#pragma unroll
    for (int mf = 0; mf < 2; ++mf)
#pragma unroll
      for (int nf = 0; nf < 4; ++nf)
        acc[mf][nf] = WMMA_BF16(aF[mf], bF[nf], acc[mf][nf]);

    // transpose-store next B tile into the alternate buffer
    if (more) {
#pragma unroll
      for (int j = 0; j < 8; ++j) Bt[1 - c][(bcs + j) * LDT + br]     = blo[j];
#pragma unroll
      for (int j = 0; j < 8; ++j) Bt[1 - c][(bcs + 8 + j) * LDT + br] = bhi[j];
      wait_async0();   // next A tile DMA landed
    }
    __syncthreads();
  }

  // C-frag layout: vgpr r, lane l -> row = r + 8*(l>=16), col = l&15
  const int rbase = (lane >> 4) << 3;
#pragma unroll
  for (int mf = 0; mf < 2; ++mf)
#pragma unroll
    for (int nf = 0; nf < 4; ++nf)
#pragma unroll
      for (int r = 0; r < 8; ++r)
        epi(bm + waveM + mf * 16 + rbase + r,
            bn + waveN + nf * 16 + ar,
            acc[mf][nf][r]);
}

// Epilogue: scatter QKV GEMM output into Q/K/V [B,H,T,hd] bf16 buffers.
struct EpiQKV {
  bf16* qkv;   // Q at 0, K at +HSZ, V at +2*HSZ
  __device__ void operator()(int row, int col, float v) const {
    const int which = col >> 10;       // 0=Q 1=K 2=V
    const int d = col & 1023;
    const int h = d >> 6;
    const int e = d & 63;
    const int b = row >> 11;
    const int t = row & 2047;
    const size_t idx = ((((size_t)b * N_HEADS + h) * SEQ) + t) * HEAD_DIM + e;
    qkv[(size_t)which * HSZ + idx] = (bf16)v;
  }
};

// Epilogue: plain fp32 row-major store (final projection).
struct EpiF32 {
  float* out;
  int ldo;
  __device__ void operator()(int row, int col, float v) const {
    out[(size_t)row * ldo + col] = v;
  }
};

// ---------------------------------------------------------------------------
// Flash-style attention. One wave per (b, h, 16-row q tile). hd = 64.
// 64 keys per chunk: 8 S-WMMAs + 8 PV-WMMAs per chunk, softmax reductions
// amortized over 64 keys. V chunk DMA'd to LDS; wait deferred past softmax.
// ---------------------------------------------------------------------------
__global__ __launch_bounds__(128)
void attn_kernel(const bf16* __restrict__ Q, const bf16* __restrict__ Kb,
                 const bf16* __restrict__ V, bf16* __restrict__ Out) {
  constexpr int KC  = 64;              // keys per chunk
  constexpr int PLD = KC + 8;          // P tile LDS stride
  constexpr int VLD = HEAD_DIM + 8;    // V tile LDS stride
  constexpr int PER_WAVE = 16 * PLD + KC * VLD;   // 5760 bf16
  __shared__ bf16 smem[4 * PER_WAVE];

  const int tid  = threadIdx.x;
  const int lane = tid & 31;
  const int wid  = tid >> 5;
  bf16* Pl = smem + wid * PER_WAVE;    // [16][PLD]
  bf16* Vl = Pl + 16 * PLD;            // [KC][VLD]

  const int g   = blockIdx.x * 4 + wid;     // 8192 waves total
  const int b   = g >> 11;
  const int rem = g & 2047;
  const int h   = rem >> 7;
  const int q0  = (rem & 127) << 4;

  const size_t headOff = (((size_t)b * N_HEADS + h) * SEQ) * HEAD_DIM;
  const bf16* Qh = Q  + headOff;
  const bf16* Kh = Kb + headOff;
  const bf16* Vh = V  + headOff;

  const int ar  = lane & 15;
  const int klo = (lane & 16) ? 8 : 0;

  // Q fragments for hd halves [0,32) and [32,64) — live in registers all loop.
  v16bf qF[2];
#pragma unroll
  for (int kk = 0; kk < 2; ++kk)
    qF[kk] = load_frag2(Qh + (size_t)(q0 + ar) * HEAD_DIM + kk * 32 + klo);

  float mrow[8], lrow[8];
  v8f o[4];
#pragma unroll
  for (int r = 0; r < 8; ++r) { mrow[r] = -1e30f; lrow[r] = 0.0f; }
#pragma unroll
  for (int nf = 0; nf < 4; ++nf)
#pragma unroll
    for (int j = 0; j < 8; ++j) o[nf][j] = 0.0f;

  for (int kc = 0; kc < SEQ; kc += KC) {
    // previous chunk's LDS gathers fully drained before DMA overwrites Vl
    wait_ds0();
    // async-DMA V rows [kc, kc+64) into per-wave LDS (2 rows per lane)
#pragma unroll
    for (int rr = 0; rr < 2; ++rr) {
      const bf16* src = Vh + (size_t)(kc + rr * 32 + lane) * HEAD_DIM;
      bf16* dst = &Vl[(rr * 32 + lane) * VLD];
#pragma unroll
      for (int j = 0; j < 8; ++j)
        async_load_b128(dst + j * 8, src + j * 8);
    }

    // S = Q K^T for 64 keys: four 16x16 accumulators over 2 hd-halves
    v8f s[4];
#pragma unroll
    for (int nt = 0; nt < 4; ++nt)
#pragma unroll
      for (int j = 0; j < 8; ++j) s[nt][j] = 0.0f;
#pragma unroll
    for (int kk = 0; kk < 2; ++kk)
#pragma unroll
      for (int nt = 0; nt < 4; ++nt) {
        v16bf kf = load_frag2(Kh + (size_t)(kc + nt * 16 + ar) * HEAD_DIM + kk * 32 + klo);
        s[nt] = WMMA_BF16(qF[kk], kf, s[nt]);
      }

    // online softmax per row over 64 keys; write P (bf16) into per-wave LDS
#pragma unroll
    for (int r = 0; r < 8; ++r) {
      float mx = fmaxf(fmaxf(s[0][r], s[1][r]), fmaxf(s[2][r], s[3][r]));
#pragma unroll
      for (int m = 8; m >= 1; m >>= 1)           // reduce within 16-lane half
        mx = fmaxf(mx, __shfl_xor(mx, m, 32));
      const float mnew  = fmaxf(mrow[r], mx);
      const float scale = __expf(mrow[r] - mnew);
      mrow[r] = mnew;
      float p[4], ps = 0.0f;
#pragma unroll
      for (int nt = 0; nt < 4; ++nt) { p[nt] = __expf(s[nt][r] - mnew); ps += p[nt]; }
#pragma unroll
      for (int m = 8; m >= 1; m >>= 1)
        ps += __shfl_xor(ps, m, 32);
      lrow[r] = lrow[r] * scale + ps;
#pragma unroll
      for (int nf = 0; nf < 4; ++nf) o[nf][r] *= scale;
      const int prow = r + ((lane >> 4) << 3);
#pragma unroll
      for (int nt = 0; nt < 4; ++nt)
        Pl[prow * PLD + nt * 16 + ar] = (bf16)p[nt];
    }

    // re-fragment P as two WMMA A-matrices (per-wave LDS; DS in-order per wave)
    v16bf pF[2];
#pragma unroll
    for (int kk = 0; kk < 2; ++kk)
      pF[kk] = load_frag2(&Pl[ar * PLD + kk * 32 + klo]);

    wait_async0();   // V DMA landed (was in flight under S-WMMA + softmax)

    // V fragments (B-matrix) gathered column-wise from LDS, then O += P @ V
#pragma unroll
    for (int nf = 0; nf < 4; ++nf) {
      const int cn = nf * 16 + ar;
#pragma unroll
      for (int kk = 0; kk < 2; ++kk) {
        v16bf vF;
#pragma unroll
        for (int j = 0; j < 8; ++j) vF[j]     = Vl[(kk * 32 + klo + j) * VLD + cn];
#pragma unroll
        for (int j = 0; j < 8; ++j) vF[8 + j] = Vl[(kk * 32 + klo + 16 + j) * VLD + cn];
        o[nf] = WMMA_BF16(pF[kk], vF, o[nf]);
      }
    }
  }

  // normalize and store attention output as bf16 [B, T, D]
#pragma unroll
  for (int nf = 0; nf < 4; ++nf)
#pragma unroll
    for (int r = 0; r < 8; ++r) {
      const int row = q0 + r + ((lane >> 4) << 3);
      const int col = h * HEAD_DIM + nf * 16 + ar;
      Out[((size_t)b * SEQ + row) * D_MODEL + col] = (bf16)(o[nf][r] / lrow[r]);
    }
}

// ---------------------------------------------------------------------------
extern "C" void kernel_launch(void* const* d_in, const int* in_sizes, int n_in,
                              void* d_out, int out_size, void* d_ws, size_t ws_size,
                              hipStream_t stream) {
  const float* x     = (const float*)d_in[0];   // [4,2048,1024]
  const float* wqkv  = (const float*)d_in[1];   // [1024,3072]
  const float* wproj = (const float*)d_in[2];   // [1024,1024]
  float* out = (float*)d_out;                    // [4,2048,1024] fp32

  // workspace layout (bf16)
  bf16* xb     = (bf16*)d_ws;                                    // 8192x1024
  bf16* wqkvb  = xb     + (size_t)NTOK * D_MODEL;                // 1024x3072
  bf16* wprojb = wqkvb  + (size_t)D_MODEL * 3 * D_MODEL;         // 1024x1024
  bf16* qkvb   = wprojb + (size_t)D_MODEL * D_MODEL;             // 3 x HSZ
  bf16* attnb  = qkvb   + 3 * HSZ;                               // 8192x1024

  const int nx = NTOK * D_MODEL;           // 8,388,608
  const int nq = D_MODEL * 3 * D_MODEL;    // 3,145,728
  const int np = D_MODEL * D_MODEL;        // 1,048,576
  cast_f32_bf16<<<4096, 256, 0, stream>>>(x,     xb,     nx);
  cast_f32_bf16<<<2048, 256, 0, stream>>>(wqkv,  wqkvb,  nq);
  cast_f32_bf16<<<1024, 256, 0, stream>>>(wproj, wprojb, np);

  // QKV projection: [8192,1024] x [1024,3072] -> scattered Q/K/V
  gemm_bf16_128x128<EpiQKV><<<(NTOK / 128) * (3 * D_MODEL / 128), 256, 0, stream>>>(
      xb, wqkvb, NTOK, 3 * D_MODEL, D_MODEL, EpiQKV{qkvb});

  // attention: 8192 q-tiles, 4 waves per block
  attn_kernel<<<2048, 128, 0, stream>>>(qkvb, qkvb + HSZ, qkvb + 2 * HSZ, attnb);

  // output projection: [8192,1024] x [1024,1024] -> fp32 out
  gemm_bf16_128x128<EpiF32><<<(NTOK / 128) * (D_MODEL / 128), 256, 0, stream>>>(
      attnb, wprojb, NTOK, D_MODEL, D_MODEL, EpiF32{out, D_MODEL});
}
